// ConvSNN_66090956751435
// MI455X (gfx1250) — matrix operand order repaired and checked by reading, not compiled, not run on hardware
//
#include <hip/hip_runtime.h>
#include <hip/hip_bf16.h>
#include <math.h>

typedef __attribute__((ext_vector_type(16))) _Float16 v16h;
typedef __attribute__((ext_vector_type(8)))  _Float16 h8;
typedef __attribute__((ext_vector_type(8)))  float    v8f;

#define EPSN 1e-5f
#define SPB 2

__device__ __forceinline__ v8f wmma_f16(v16h a, v16h b, v8f c) {
  return __builtin_amdgcn_wmma_f32_16x16x32_f16(false, a, false, b, (short)0, c,
                                                false, false);
}

// B fragment: lane holds col n; element e needs k = (lane>=16?16:0)+e, i.e. 16
// consecutive halves of one [n][Kpad] row -> 2 x b128 loads.
__device__ __forceinline__ v16h load_b16(const _Float16* p) {
  v16h r;
  ((h8*)&r)[0] = *(const h8*)(p);
  ((h8*)&r)[1] = *(const h8*)(p + 8);
  return r;
}

// Row-major A fragment: 16 contiguous halves at k = 2*kb8 -> 2 x b128 loads.
__device__ __forceinline__ v16h load_a_rm(const _Float16* row, int kb8) {
  v16h r;
  ((h8*)&r)[0] = *(const h8*)(row + 2 * kb8);
  ((h8*)&r)[1] = *(const h8*)(row + 2 * kb8 + 8);
  return r;
}

// Implicit-GEMM conv A fragment from a zero-padded LDS image.
// base = plane0 + window_top_left ; OFF(k) = (k/9)*ps + ((k%9)/3)*rs + (k%9)%3.
// ks/kmax/ps/rs are compile-time at every call site, so div/mod folds; the
// lane-half pick is a single cndmask of two constants.
__device__ __forceinline__ v16h conv_a(const _Float16* base, int hi, int ks,
                                       int kmax, int ps, int rs) {
  v16h r;
#pragma unroll
  for (int e = 0; e < 16; ++e) {
    int k0 = ks * 32 + e;  // lo-half lanes: k = e
    int k1 = k0 + 16;      // hi-half lanes: k = 16 + e
    int c0 = k0 / 9, r0 = k0 - c0 * 9;
    int c1 = k1 / 9, r1 = k1 - c1 * 9;
    int off0 = c0 * ps + (r0 / 3) * rs + (r0 % 3);
    int off1 = c1 * ps + (r1 / 3) * rs + (r1 % 3);
    _Float16 v;
    if (k1 < kmax)      v = base[hi ? off1 : off0];
    else if (k0 < kmax) v = hi ? (_Float16)0.f : base[off0];
    else                v = (_Float16)0.f;
    r[e] = v;
  }
  return r;
}

// ---------------------------------------------------------------------------
// Kernel 0: convert all weights to k-padded f16 tables in workspace.
// Regions (halves): w1p@0[16x64] w2p@1024[32x160] w3p@6144[64x288]
// fcp@24576[256x1056] wihp@294912[768x256] whhp@491520[768x256]  (end 688128)
// ---------------------------------------------------------------------------
__global__ __launch_bounds__(256) void prep_weights(
    const float* __restrict__ w1, const float* __restrict__ w2,
    const float* __restrict__ w3, const float* __restrict__ fcw,
    const float* __restrict__ wih, const float* __restrict__ whh,
    _Float16* __restrict__ wt) {
  int i = blockIdx.x * 256 + threadIdx.x;
  if (i < 1024) {
    int n = i >> 6, k = i & 63;
    wt[i] = (_Float16)(k < 36 ? w1[n * 36 + k] : 0.f);
  } else if (i < 6144) {
    int j = i - 1024, n = j / 160, k = j - n * 160;
    wt[i] = (_Float16)(k < 144 ? w2[n * 144 + k] : 0.f);
  } else if (i < 24576) {
    wt[i] = (_Float16)w3[i - 6144];
  } else if (i < 294912) {
    int j = i - 24576, n = j / 1056, k = j - n * 1056;
    wt[i] = (_Float16)(k < 1033 ? fcw[n * 1033 + k] : 0.f);
  } else if (i < 491520) {
    wt[i] = (_Float16)wih[i - 294912];
  } else if (i < 688128) {
    wt[i] = (_Float16)whh[i - 491520];
  }
}

// ---------------------------------------------------------------------------
// Kernel 1: conv1+GN+relu -> conv2+GN+relu -> conv3+GN+relu, LDS-resident with
// zero-padded activation planes. One wave per sample, SPB waves per block.
// ---------------------------------------------------------------------------
__global__ __launch_bounds__(64) void conv_stack_kernel(
    const float* __restrict__ x,
    const _Float16* __restrict__ w1p, const float* __restrict__ b1,
    const float* __restrict__ g1w, const float* __restrict__ g1b,
    const _Float16* __restrict__ w2p, const float* __restrict__ b2,
    const float* __restrict__ g2w, const float* __restrict__ g2b,
    const _Float16* __restrict__ w3p, const float* __restrict__ b3,
    const float* __restrict__ g3w, const float* __restrict__ g3b,
    _Float16* __restrict__ c3out) {
  __shared__ __align__(16) _Float16 xp[SPB][1160];    // 4 x 17 x 17 (padded)
  __shared__ __align__(16) _Float16 c1lin[SPB][3600]; // conv1 raw, [ch][225]
  __shared__ __align__(16) _Float16 c1p[SPB][4624];   // 16 x 17 x 17 (padded)
  __shared__ __align__(16) _Float16 c2p[SPB][3200];   // 32 x 10 x 10 (padded)
  __shared__ float ps_[SPB][4][32];
  __shared__ float ps2_[SPB][4][32];
  __shared__ float gsum[SPB][8], gsum2[SPB][8];

  const int wave = threadIdx.x >> 5;
  const int lane = threadIdx.x & 31;
  const long long s = (long long)blockIdx.x * SPB + wave;
  const int nn = lane & 15;
  const int hi = lane >> 4;
  const int kb16 = hi ? 16 : 0;
  const int mb = hi ? 8 : 0;

  {  // zero padded planes
    unsigned int* p0 = (unsigned int*)xp[wave];
    for (int i = lane; i < 580; i += 32) p0[i] = 0u;
    unsigned int* p1 = (unsigned int*)c1p[wave];
    for (int i = lane; i < 2312; i += 32) p1[i] = 0u;
    unsigned int* p2 = (unsigned int*)c2p[wave];
    for (int i = lane; i < 1600; i += 32) p2[i] = 0u;
  }
  __syncthreads();
  for (int i = lane; i < 900; i += 32) {  // fill padded input
    int ci = i / 225, p = i - ci * 225;
    int y = p / 15, xx = p - y * 15;
    xp[wave][ci * 289 + (y + 1) * 17 + (xx + 1)] = (_Float16)x[s * 900 + i];
  }
  __syncthreads();

  // ---- conv1: M=225(15 tiles) x N=16 x K=36(pad64) ----
  v16h b1f0 = load_b16(w1p + nn * 64 + kb16);
  v16h b1f1 = load_b16(w1p + nn * 64 + 32 + kb16);
  auto c1tile = [&](int mt) -> v8f {
    int m0 = mt * 16 + nn;
    int m2 = m0 < 225 ? m0 : 224;
    int oy = m2 / 15, ox = m2 - oy * 15;
    const _Float16* base = &xp[wave][oy * 17 + ox];
    v8f c = {};
#pragma unroll
    for (int ks = 0; ks < 2; ++ks) {
      v16h a = conv_a(base, hi, ks, 36, 289, 17);
      c = wmma_f16(a, ks ? b1f1 : b1f0, c);
    }
    return c;
  };
  {
    float p = 0.f, p2 = 0.f;
    float bias = b1[nn];
    _Float16* crow = &c1lin[wave][nn * 225 + mb];
    for (int mt = 0; mt < 14; ++mt) {  // all 16 rows valid; linear store, no div
      v8f c = c1tile(mt);
#pragma unroll
      for (int v = 0; v < 8; ++v) {
        float val = c[v] + bias;
        p += val;
        p2 += val * val;
        crow[mt * 16 + v] = (_Float16)val;
      }
    }
    {  // tail tile mt=14: only m=224 (v=0, lo lanes) is valid
      v8f c = c1tile(14);
      if (!hi) {
        float val = c[0] + bias;
        p += val;
        p2 += val * val;
        c1lin[wave][nn * 225 + 224] = (_Float16)val;
      }
    }
    ps_[wave][0][lane] = p;
    ps2_[wave][0][lane] = p2;
  }
  __syncthreads();
  if (lane < 4) {  // 4 groups x 4 channels
    float a0 = 0.f, a1 = 0.f;
    for (int c = lane * 4; c < lane * 4 + 4; ++c) {
      a0 += ps_[wave][0][c] + ps_[wave][0][16 + c];
      a1 += ps2_[wave][0][c] + ps2_[wave][0][16 + c];
    }
    gsum[wave][lane] = a0;
    gsum2[wave][lane] = a1;
  }
  __syncthreads();
  for (int r = lane; r < 240; r += 32) {  // GN1: linear -> padded, row-wise
    int ch = r / 15, y = r - ch * 15;
    int g = ch >> 2;
    float mu = gsum[wave][g] * (1.f / 900.f);
    float rstd = rsqrtf(gsum2[wave][g] * (1.f / 900.f) - mu * mu + EPSN);
    float gw = g1w[ch], gb = g1b[ch];
    const _Float16* src = &c1lin[wave][ch * 225 + y * 15];
    _Float16* dst = &c1p[wave][ch * 289 + (y + 1) * 17 + 1];
#pragma unroll
    for (int xx = 0; xx < 15; ++xx) {
      float val = ((float)src[xx] - mu) * rstd * gw + gb;
      dst[xx] = (_Float16)fmaxf(val, 0.f);
    }
  }
  __syncthreads();

  // ---- conv2: M=64(4) x N=32(2) x K=144(pad160) ----
  {
    float pA0 = 0.f, pA1 = 0.f, pB0 = 0.f, pB1 = 0.f;
    for (int mt = 0; mt < 4; ++mt) {
      int m0 = mt * 16 + nn;
      int oy = m0 >> 3, ox = m0 & 7;
      const _Float16* base = &c1p[wave][(oy * 2) * 17 + ox * 2];
      v16h af[5];
#pragma unroll
      for (int ks = 0; ks < 5; ++ks) af[ks] = conv_a(base, hi, ks, 144, 289, 17);
#pragma unroll
      for (int nt = 0; nt < 2; ++nt) {
        v8f c = {};
#pragma unroll
        for (int ks = 0; ks < 5; ++ks)
          c = wmma_f16(af[ks], load_b16(w2p + (nt * 16 + nn) * 160 + ks * 32 + kb16), c);
        int ch = nt * 16 + nn;
        float bias = b2[ch];
        float lp = 0.f, lp2 = 0.f;
#pragma unroll
        for (int v = 0; v < 8; ++v) {
          int m = mt * 16 + v + mb;
          float val = c[v] + bias;
          lp += val;
          lp2 += val * val;
          int yy = m >> 3, xx = m & 7;
          c2p[wave][ch * 100 + (yy + 1) * 10 + (xx + 1)] = (_Float16)val;
        }
        if (nt == 0) { pA0 += lp; pB0 += lp2; } else { pA1 += lp; pB1 += lp2; }
      }
    }
    ps_[wave][0][lane] = pA0;
    ps_[wave][1][lane] = pA1;
    ps2_[wave][0][lane] = pB0;
    ps2_[wave][1][lane] = pB1;
  }
  __syncthreads();
  if (lane < 8) {  // 8 groups x 4 channels
    float a0 = 0.f, a1 = 0.f;
    for (int c = lane * 4; c < lane * 4 + 4; ++c) {
      int slot = c >> 4, ll = c & 15;
      a0 += ps_[wave][slot][ll] + ps_[wave][slot][16 + ll];
      a1 += ps2_[wave][slot][ll] + ps2_[wave][slot][16 + ll];
    }
    gsum[wave][lane] = a0;
    gsum2[wave][lane] = a1;
  }
  __syncthreads();
  for (int i = lane; i < 2048; i += 32) {  // GN2 normalize (shift-only idx math)
    int ch = i >> 6, p = i & 63;
    int idx = ch * 100 + ((p >> 3) + 1) * 10 + (p & 7) + 1;
    float mu = gsum[wave][ch >> 2] * (1.f / 256.f);
    float var = gsum2[wave][ch >> 2] * (1.f / 256.f) - mu * mu;
    float val = ((float)c2p[wave][idx] - mu) * rsqrtf(var + EPSN) * g2w[ch] + g2b[ch];
    c2p[wave][idx] = (_Float16)fmaxf(val, 0.f);
  }
  __syncthreads();

  // ---- conv3: M=16(1) x N=64(4) x K=288 ----
  v8f acc[4] = {};
  {
    int oy = nn >> 2, ox = nn & 3;
    const _Float16* base = &c2p[wave][(oy * 2) * 10 + ox * 2];
#pragma unroll
    for (int ks = 0; ks < 9; ++ks) {
      v16h a = conv_a(base, hi, ks, 288, 100, 10);
#pragma unroll
      for (int nt = 0; nt < 4; ++nt)
        acc[nt] = wmma_f16(a, load_b16(w3p + (nt * 16 + nn) * 288 + ks * 32 + kb16), acc[nt]);
    }
  }
#pragma unroll
  for (int nt = 0; nt < 4; ++nt) {
    int ch = nt * 16 + nn;
    float bias = b3[ch];
    float p = 0.f, p2 = 0.f;
#pragma unroll
    for (int v = 0; v < 8; ++v) {
      float val = acc[nt][v] + bias;
      p += val;
      p2 += val * val;
    }
    ps_[wave][nt][lane] = p;
    ps2_[wave][nt][lane] = p2;
  }
  __syncthreads();
  if (lane < 8) {  // 8 groups x 8 channels
    float a0 = 0.f, a1 = 0.f;
    for (int c = lane * 8; c < lane * 8 + 8; ++c) {
      int slot = c >> 4, ll = c & 15;
      a0 += ps_[wave][slot][ll] + ps_[wave][slot][16 + ll];
      a1 += ps2_[wave][slot][ll] + ps2_[wave][slot][16 + ll];
    }
    gsum[wave][lane] = a0;
    gsum2[wave][lane] = a1;
  }
  __syncthreads();
#pragma unroll
  for (int nt = 0; nt < 4; ++nt) {
    int ch = nt * 16 + nn, g = ch >> 3;
    float mu = gsum[wave][g] * (1.f / 128.f);
    float rs = rsqrtf(gsum2[wave][g] * (1.f / 128.f) - mu * mu + EPSN);
    float bias = b3[ch], gw = g3w[ch], gb = g3b[ch];
#pragma unroll
    for (int v = 0; v < 8; ++v) {
      int m = v + mb;
      float val = (acc[nt][v] + bias - mu) * rs * gw + gb;
      c3out[s * 1024 + ch * 16 + m] = (_Float16)fmaxf(val, 0.f);
    }
  }
}

// ---------------------------------------------------------------------------
// Kernel 2: fc + LN1 + relu -> GRU step -> LN2 -> actor/critic heads.
// One block = 16-sample tile, 4 waves. All A loads b128 from LDS, all B loads
// b128 from prepped f16 weight tables.
// ---------------------------------------------------------------------------
__global__ __launch_bounds__(128) void head_kernel(
    const _Float16* __restrict__ c3f, const float* __restrict__ scal,
    const float* __restrict__ hidden,
    const _Float16* __restrict__ fcp, const float* __restrict__ fcb,
    const float* __restrict__ ln1w, const float* __restrict__ ln1b,
    const _Float16* __restrict__ wihp, const _Float16* __restrict__ whhp,
    const float* __restrict__ bih, const float* __restrict__ bhh,
    const float* __restrict__ ln2w, const float* __restrict__ ln2b,
    const float* __restrict__ aw, const float* __restrict__ ab,
    const float* __restrict__ cw, const float* __restrict__ cb,
    float* __restrict__ out, int nB) {
  __shared__ __align__(16) char smem[33792 + 16384];
  __shared__ float rmu[16], rsig[16];
  _Float16* ah = (_Float16*)smem;           // phase 1: 16 x 1056 f16 A tile
  float* ef32 = (float*)(smem + 33792);     // 16 x 256 f32 (fc out / later feat)
  _Float16* eh = (_Float16*)smem;           // phase 2: 16 x 256 f16 embed
  _Float16* hh = (_Float16*)(smem + 8192);  // phase 2: 16 x 256 f16 hidden
  float* feat = ef32;                       // phase 3: h_new f32

  const int tid = threadIdx.x;
  const int wave = tid >> 5, lane = tid & 31;
  const int nn = lane & 15;
  const int kb8 = (lane >= 16) ? 8 : 0;
  const int kb16 = (lane >= 16) ? 16 : 0;
  const int mb = (lane >= 16) ? 8 : 0;
  const long long s0 = (long long)blockIdx.x * 16;

  for (int m = 0; m < 16; ++m) {  // stage A tile: c3 flat | scalars | zero pad
    long long s = s0 + m;
    for (int k = tid; k < 1056; k += 128) {
      _Float16 v;
      if (k < 1024)      v = c3f[s * 1024 + k];
      else if (k < 1033) v = (_Float16)scal[s * 9 + (k - 1024)];
      else               v = (_Float16)0.f;
      ah[m * 1056 + k] = v;
    }
  }
  __syncthreads();

  // fc: M=16 x N=256 (wave owns n-tiles {w,4+w,8+w,12+w}) x K=1056 (33 ksteps)
  const _Float16* arow = ah + nn * 1056;
  for (int jj = 0; jj < 4; ++jj) {
    int nt = wave + 4 * jj;
    const _Float16* brow = fcp + (nt * 16 + nn) * 1056 + kb16;
    v8f c = {};
    for (int ks = 0; ks < 33; ++ks)
      c = wmma_f16(load_a_rm(arow + ks * 32, kb8), load_b16(brow + ks * 32), c);
    int col = nt * 16 + nn;
    float bias = fcb[col];
#pragma unroll
    for (int v = 0; v < 8; ++v) ef32[(v + mb) * 256 + col] = c[v] + bias;
  }
  __syncthreads();
  if (tid < 16) {  // LN1 row stats
    float su = 0.f, s2 = 0.f;
    for (int c = 0; c < 256; ++c) {
      float v = ef32[tid * 256 + c];
      su += v;
      s2 += v * v;
    }
    float mu = su * (1.f / 256.f);
    rmu[tid] = mu;
    rsig[tid] = rsqrtf(s2 * (1.f / 256.f) - mu * mu + EPSN);
  }
  __syncthreads();
  for (int i = tid; i < 4096; i += 128) {  // embed f16 + hidden f16 tiles
    int m = i >> 8, c = i & 255;
    float v = (ef32[i] - rmu[m]) * rsig[m] * ln1w[c] + ln1b[c];
    eh[i] = (_Float16)fmaxf(v, 0.f);
    hh[i] = (_Float16)hidden[(s0 + m) * 256 + c];
  }
  __syncthreads();

  // GRU: wave owns matching r/z/n column tiles j -> gates fuse in registers
  const _Float16* erow = eh + nn * 256;
  const _Float16* hrow = hh + nn * 256;
  for (int jj = 0; jj < 4; ++jj) {
    int j = wave + 4 * jj;
    const _Float16* br = wihp + (j * 16 + nn) * 256 + kb16;
    const _Float16* bhr = whhp + (j * 16 + nn) * 256 + kb16;
    const _Float16* bz = br + 256 * 256;
    const _Float16* bhz = bhr + 256 * 256;
    const _Float16* bn = br + 512 * 256;
    const _Float16* bhn = bhr + 512 * 256;
    v8f ir = {}, hr = {}, iz = {}, hz = {}, inn = {}, hn = {};
    for (int ks = 0; ks < 8; ++ks) {
      v16h ae = load_a_rm(erow + ks * 32, kb8);
      v16h ahd = load_a_rm(hrow + ks * 32, kb8);
      ir = wmma_f16(ae, load_b16(br + ks * 32), ir);
      hr = wmma_f16(ahd, load_b16(bhr + ks * 32), hr);
      iz = wmma_f16(ae, load_b16(bz + ks * 32), iz);
      hz = wmma_f16(ahd, load_b16(bhz + ks * 32), hz);
      inn = wmma_f16(ae, load_b16(bn + ks * 32), inn);
      hn = wmma_f16(ahd, load_b16(bhn + ks * 32), hn);
    }
    int col = j * 16 + nn;
    float br_ = bih[col] + bhh[col];
    float bz_ = bih[256 + col] + bhh[256 + col];
    float bin = bih[512 + col], bhn_ = bhh[512 + col];
#pragma unroll
    for (int v = 0; v < 8; ++v) {
      int m = v + mb;
      long long s = s0 + m;
      float r = 1.f / (1.f + expf(-(ir[v] + hr[v] + br_)));
      float z = 1.f / (1.f + expf(-(iz[v] + hz[v] + bz_)));
      float nv = tanhf(inn[v] + bin + r * (hn[v] + bhn_));
      float hprev = hidden[s * 256 + col];
      float hnew = (1.f - z) * nv + z * hprev;
      out[(long long)nB * 10 + s * 256 + col] = hnew;
      feat[m * 256 + col] = hnew;
    }
  }
  __syncthreads();
  if (tid < 16) {  // LN2 row stats
    float su = 0.f, s2 = 0.f;
    for (int c = 0; c < 256; ++c) {
      float v = feat[tid * 256 + c];
      su += v;
      s2 += v * v;
    }
    float mu = su * (1.f / 256.f);
    rmu[tid] = mu;
    rsig[tid] = rsqrtf(s2 * (1.f / 256.f) - mu * mu + EPSN);
  }
  __syncthreads();
  for (int i = tid; i < 144; i += 128) {  // actor head: 16 x 9
    int m = i / 9, a = i - m * 9;
    long long s = s0 + m;
    float acc = ab[a];
    for (int c = 0; c < 256; ++c) {
      float fv = (feat[m * 256 + c] - rmu[m]) * rsig[m] * ln2w[c] + ln2b[c];
      acc += fv * aw[a * 256 + c];
    }
    out[s * 9 + a] = acc;
  }
  if (tid < 16) {  // critic head: 16 x 1
    int m = tid;
    long long s = s0 + m;
    float acc = cb[0];
    for (int c = 0; c < 256; ++c) {
      float fv = (feat[m * 256 + c] - rmu[m]) * rsig[m] * ln2w[c] + ln2b[c];
      acc += fv * cw[c];
    }
    out[(long long)nB * 9 + s] = acc;
  }
}

extern "C" void kernel_launch(void* const* d_in, const int* in_sizes, int n_in,
                              void* d_out, int out_size, void* d_ws, size_t ws_size,
                              hipStream_t stream) {
  (void)n_in;
  (void)out_size;
  (void)ws_size;
  const float* x = (const float*)d_in[0];
  const float* scal = (const float*)d_in[1];
  const float* hid = (const float*)d_in[2];
  const float* w1 = (const float*)d_in[3];
  const float* b1 = (const float*)d_in[4];
  const float* g1w = (const float*)d_in[5];
  const float* g1b = (const float*)d_in[6];
  const float* w2 = (const float*)d_in[7];
  const float* b2 = (const float*)d_in[8];
  const float* g2w = (const float*)d_in[9];
  const float* g2b = (const float*)d_in[10];
  const float* w3 = (const float*)d_in[11];
  const float* b3 = (const float*)d_in[12];
  const float* g3w = (const float*)d_in[13];
  const float* g3b = (const float*)d_in[14];
  const float* fcw = (const float*)d_in[15];
  const float* fcb = (const float*)d_in[16];
  const float* l1w = (const float*)d_in[17];
  const float* l1b = (const float*)d_in[18];
  const float* wih = (const float*)d_in[19];
  const float* whh = (const float*)d_in[20];
  const float* bih = (const float*)d_in[21];
  const float* bhh = (const float*)d_in[22];
  const float* l2w = (const float*)d_in[23];
  const float* l2b = (const float*)d_in[24];
  const float* aw = (const float*)d_in[25];
  const float* ab = (const float*)d_in[26];
  const float* cw = (const float*)d_in[27];
  const float* cb = (const float*)d_in[28];
  float* out = (float*)d_out;

  const int nB = in_sizes[0] / 900;  // B = 32768
  _Float16* wsh = (_Float16*)d_ws;
  _Float16* c3f = wsh;                     // nB x 1024 f16
  _Float16* wt = wsh + (size_t)nB * 1024;  // 688128 halves of weight tables
  _Float16* w1p = wt;
  _Float16* w2p = wt + 1024;
  _Float16* w3p = wt + 6144;
  _Float16* fcp = wt + 24576;
  _Float16* wihp = wt + 294912;
  _Float16* whhp = wt + 491520;

  prep_weights<<<2688, 256, 0, stream>>>(w1, w2, w3, fcw, wih, whh, wt);
  conv_stack_kernel<<<nB / SPB, 64, 0, stream>>>(
      x, w1p, b1, g1w, g1b, w2p, b2, g2w, g2b, w3p, b3, g3w, g3b, c3f);
  head_kernel<<<nB / 16, 128, 0, stream>>>(c3f, scal, hid, fcp, fcb, l1w, l1b,
                                           wihp, whhp, bih, bhh, l2w, l2b, aw,
                                           ab, cw, cb, out, nB);
}
